// ScaleDotProductAttention_79104707657949
// MI455X (gfx1250) — matrix operand order repaired and checked by reading, not compiled, not run on hardware
//
#include <hip/hip_runtime.h>

// ---------------------------------------------------------------------------
// Flash attention for [B=2,H=16,S=2048,D=64] fp32, unscaled scores (matches
// reference). bf16 WMMA (v_wmma_f32_16x16x32_bf16) for both GEMMs, f32
// accumulate, online softmax. Software-pipelined with double-buffered LDS
// K/V tiles; softmax row reductions done with DPP16 (VALU) instead of
// ds_bpermute so they co-execute with the XDL WMMA pipe. wave32 / gfx1250.
// ---------------------------------------------------------------------------

typedef __attribute__((ext_vector_type(16))) __bf16 v16bf;
typedef __attribute__((ext_vector_type(8)))  float  v8f;

#define S_LEN 2048
#define DHEAD 64
#define BM    128          // query rows per workgroup (8 waves x 16)
#define BN    64           // key rows per LDS tile
#define NT    (S_LEN / BN) // 32 key tiles
#define LDK   72           // padded LDS row length in bf16 elems (144B = 9*16B)

struct bf16x16_bits { uint4 a, b; };

// 32 contiguous bytes -> one 16-element bf16 fragment (2x ds_load_b128)
static __device__ __forceinline__ v16bf ld_frag(const __bf16* p) {
    bf16x16_bits t;
    t.a = *(const uint4*)(p);
    t.b = *(const uint4*)(p + 8);
    return __builtin_bit_cast(v16bf, t);
}
// two disjoint 16-byte runs (A-matrix layout)
static __device__ __forceinline__ v16bf ld_frag_split(const __bf16* p0, const __bf16* p1) {
    bf16x16_bits t;
    t.a = *(const uint4*)(p0);
    t.b = *(const uint4*)(p1);
    return __builtin_bit_cast(v16bf, t);
}

// ---- DPP16 cross-lane reductions within each 16-lane row ------------------
// 0xB1 = quad_perm(1,0,3,2) : lane^1        0x4E = quad_perm(2,3,0,1) : lane^2
// 0x141 = row_half_mirror   : lane^7        0x140 = row_mirror        : lane^15
template <int CTRL>
static __device__ __forceinline__ float dpp_xchg(float x) {
    return __builtin_bit_cast(float,
        __builtin_amdgcn_mov_dpp(__builtin_bit_cast(int, x), CTRL, 0xF, 0xF, true));
}
static __device__ __forceinline__ float row_max16(float x) {
    x = fmaxf(x, dpp_xchg<0xB1>(x));
    x = fmaxf(x, dpp_xchg<0x4E>(x));
    x = fmaxf(x, dpp_xchg<0x141>(x));
    x = fmaxf(x, dpp_xchg<0x140>(x));
    return x;
}
static __device__ __forceinline__ float row_sum16(float x) {
    x += dpp_xchg<0xB1>(x);
    x += dpp_xchg<0x4E>(x);
    x += dpp_xchg<0x141>(x);
    x += dpp_xchg<0x140>(x);
    return x;
}

// issue the 128B of global loads for this thread's share of a K/V tile
static __device__ __forceinline__ void load_tile_regs(
    const float* __restrict__ kbase, const float* __restrict__ vbase, int kt,
    int sk_n, int sk_d0, int sv_n, int sv_d0, float* kr, float* vr)
{
    const float4* kp = (const float4*)(kbase + (size_t)(kt + sk_n) * DHEAD + sk_d0);
    const float4* vp = (const float4*)(vbase + (size_t)(kt + sv_n) * DHEAD + sv_d0);
#pragma unroll
    for (int e = 0; e < 4; ++e) {
        float4 a = kp[e];
        kr[e * 4 + 0] = a.x; kr[e * 4 + 1] = a.y; kr[e * 4 + 2] = a.z; kr[e * 4 + 3] = a.w;
        float4 b = vp[e];
        vr[e * 4 + 0] = b.x; vr[e * 4 + 1] = b.y; vr[e * 4 + 2] = b.z; vr[e * 4 + 3] = b.w;
    }
}

// convert + commit a staged tile into an LDS buffer (K row-major, V transposed)
static __device__ __forceinline__ void store_tile_lds(
    __bf16 (*Kld)[LDK], __bf16 (*Vtd)[LDK],
    int sk_n, int sk_d0, int sv_n, int sv_d0,
    const float* kr, const float* vr)
{
#pragma unroll
    for (int i = 0; i < 16; ++i) Kld[sk_n][sk_d0 + i] = (__bf16)kr[i];
#pragma unroll
    for (int i = 0; i < 16; ++i) Vtd[sv_d0 + i][sv_n] = (__bf16)vr[i];
}

__global__ __launch_bounds__(256)
void fa_fwd_kernel(const float* __restrict__ Q, const float* __restrict__ K,
                   const float* __restrict__ V, float* __restrict__ O)
{
    __shared__ __align__(16) __bf16 Kl[2][BN][LDK];     // K tiles, row-major [key][d]
    __shared__ __align__(16) __bf16 Vt[2][DHEAD][LDK];  // V tiles, transposed [d][key]
    __shared__ __align__(16) __bf16 Pl[8][16][LDK];     // per-wave P scratch [m][key]

    const int tid  = threadIdx.x;
    const int wave = tid >> 5;
    const int lane = tid & 31;
    const int lrow = lane & 15;      // M (A) / N (B,C) index
    const int hl   = lane >> 4;      // lane-half selector

    const int bh = blockIdx.y;       // fused batch*head, 0..31
    const int qt = blockIdx.x;       // query tile, 0..15
    const size_t base = (size_t)bh * S_LEN * DHEAD;
    const float* kbase = K + base;
    const float* vbase = V + base;

    // ---- Q fragments: A-layout 16x32 bf16, two K-chunks cover D=64 --------
    const int qrow = qt * BM + wave * 16 + lrow;
    const float* qp = Q + base + (size_t)qrow * DHEAD;
    v16bf aq[2];
#pragma unroll
    for (int c = 0; c < 2; ++c) {
#pragma unroll
        for (int i = 0; i < 8; ++i) {
            aq[c][i]     = (__bf16)qp[c * 32 + hl * 8 + i];
            aq[c][i + 8] = (__bf16)qp[c * 32 + hl * 8 + 16 + i];
        }
    }

    v8f   o_acc[4] = {};                 // 16x64 output accumulator (f32)
    float m_i[8], l_i[8];
#pragma unroll
    for (int r = 0; r < 8; ++r) { m_i[r] = -3.0e38f; l_i[r] = 0.f; }

    // staging thread mapping (256 threads, 64x64 tiles, 16 elems/thread)
    const int sk_n  = tid >> 2;              // key row for Kl
    const int sk_d0 = (tid & 3) << 4;        // d chunk start
    const int sv_n  = tid & 63;              // key row for Vt (transposed write)
    const int sv_d0 = (tid >> 6) << 4;

    // ---- prologue: stage tile 0 into buffer 0 -----------------------------
    {
        float kr[16], vr[16];
        load_tile_regs(kbase, vbase, 0, sk_n, sk_d0, sv_n, sv_d0, kr, vr);
        store_tile_lds(Kl[0], Vt[0], sk_n, sk_d0, sv_n, sv_d0, kr, vr);
    }
    __syncthreads();

    for (int it = 0; it < NT; ++it) {
        const int buf = it & 1;

        // ---- issue next tile's global loads NOW (overlap with WMMAs) -----
        const int ktn = (it + 1 < NT) ? (it + 1) * BN : it * BN;   // clamp on last
        float kr[16], vr[16];
        load_tile_regs(kbase, vbase, ktn, sk_n, sk_d0, sv_n, sv_d0, kr, vr);
        if (it + 2 < NT) {               // global_prefetch_b8 of tile after next
            __builtin_prefetch(kbase + (size_t)((it + 2) * BN + sk_n) * DHEAD + sk_d0, 0, 0);
            __builtin_prefetch(vbase + (size_t)((it + 2) * BN + sv_n) * DHEAD + sv_d0, 0, 0);
        }

        // ---- scores S = Q K^T : 4 subtiles of 16 keys, 2 WMMAs each ------
        v8f sc[4];
#pragma unroll
        for (int nt = 0; nt < 4; ++nt) {
            const __bf16* kp = &Kl[buf][nt * 16 + lrow][0];
            v16bf bk0 = ld_frag_split(kp + hl * 8, kp + hl * 8 + 16);             // K-chunk d=0..31
            v16bf bk1 = ld_frag_split(kp + 32 + hl * 8, kp + 32 + hl * 8 + 16);   // K-chunk d=32..63
            v8f acc = {};
            acc = __builtin_amdgcn_wmma_f32_16x16x32_bf16(false, aq[0], false, bk0,
                                                          (short)0, acc, false, false);
            acc = __builtin_amdgcn_wmma_f32_16x16x32_bf16(false, aq[1], false, bk1,
                                                          (short)0, acc, false, false);
            sc[nt] = acc;
        }

        // ---- online softmax; row reductions via DPP16 (VALU-only) --------
        float alpha[8];
#pragma unroll
        for (int r = 0; r < 8; ++r) {
            float mx = fmaxf(fmaxf(sc[0][r], sc[1][r]), fmaxf(sc[2][r], sc[3][r]));
            mx = row_max16(mx);
            float mnew = fmaxf(m_i[r], mx);
            alpha[r] = __expf(m_i[r] - mnew);
            m_i[r] = mnew;
        }
        float rsum[8];
#pragma unroll
        for (int r = 0; r < 8; ++r) rsum[r] = 0.f;
#pragma unroll
        for (int nt = 0; nt < 4; ++nt) {
#pragma unroll
            for (int r = 0; r < 8; ++r) {
                float p = __expf(sc[nt][r] - m_i[r]);
                rsum[r] += p;
                // C-layout -> row-major P scratch (wave-private)
                Pl[wave][r + hl * 8][nt * 16 + lrow] = (__bf16)p;
            }
        }
#pragma unroll
        for (int r = 0; r < 8; ++r)
            l_i[r] = l_i[r] * alpha[r] + row_sum16(rsum[r]);
#pragma unroll
        for (int nt = 0; nt < 4; ++nt)
#pragma unroll
            for (int r = 0; r < 8; ++r)
                o_acc[nt][r] *= alpha[r];

        // ---- O += P V : reload P in A-layout, V^T fragments from LDS -----
        const __bf16* pr = &Pl[wave][lrow][0];
        v16bf pa[2];
#pragma unroll
        for (int c = 0; c < 2; ++c)
            pa[c] = ld_frag_split(pr + c * 32 + hl * 8,
                                  pr + c * 32 + hl * 8 + 16);
#pragma unroll
        for (int nt = 0; nt < 4; ++nt) {
            const __bf16* vp = &Vt[buf][nt * 16 + lrow][0];
            v16bf bv0 = ld_frag_split(vp + hl * 8, vp + hl * 8 + 16);             // keys 0..31
            v16bf bv1 = ld_frag_split(vp + 32 + hl * 8, vp + 32 + hl * 8 + 16);   // keys 32..63
            o_acc[nt] = __builtin_amdgcn_wmma_f32_16x16x32_bf16(false, pa[0], false, bv0,
                                                                (short)0, o_acc[nt], false, false);
            o_acc[nt] = __builtin_amdgcn_wmma_f32_16x16x32_bf16(false, pa[1], false, bv1,
                                                                (short)0, o_acc[nt], false, false);
        }

        // ---- commit next tile into the alternate buffer, single barrier --
        store_tile_lds(Kl[buf ^ 1], Vt[buf ^ 1], sk_n, sk_d0, sv_n, sv_d0, kr, vr);
        __syncthreads();
    }

    // ---- normalize and write out (f32, coalesced 64B per row segment) ----
#pragma unroll
    for (int r = 0; r < 8; ++r) {
        float inv = 1.f / l_i[r];
        int row = qt * BM + wave * 16 + r + hl * 8;
        float* orow = O + base + (size_t)row * DHEAD;
#pragma unroll
        for (int nt = 0; nt < 4; ++nt)
            orow[nt * 16 + lrow] = o_acc[nt][r] * inv;
    }
}

extern "C" void kernel_launch(void* const* d_in, const int* in_sizes, int n_in,
                              void* d_out, int out_size, void* d_ws, size_t ws_size,
                              hipStream_t stream) {
    (void)in_sizes; (void)n_in; (void)out_size; (void)d_ws; (void)ws_size;
    const float* Q = (const float*)d_in[0];
    const float* K = (const float*)d_in[1];
    const float* V = (const float*)d_in[2];
    float* O = (float*)d_out;
    dim3 grid(S_LEN / BM, 32, 1);   // 16 query tiles x (B*H)=32
    dim3 block(256, 1, 1);          // 8 waves of 32
    hipLaunchKernelGGL(fa_fwd_kernel, grid, block, 0, stream, Q, K, V, O);
}
